// ConditionalDLFactorized15_74680891343534
// MI455X (gfx1250) — matrix operand order repaired and checked by reading, not compiled, not run on hardware
//
#include <hip/hip_runtime.h>
#include <hip/hip_bf16.h>
#include <stdint.h>

// dims
#define TB  2048          // T*B tokens
#define CIN 256
#define COUT 256
#define NE  64
#define RD  64
#define TAU 1.0f
#define COMMIT 0.25f
#define KMEANS 1.0f
#define IMP 30.0f

typedef __attribute__((ext_vector_type(16))) __bf16 v16bf;
typedef __attribute__((ext_vector_type(8)))  float  v8f;
typedef __attribute__((ext_vector_type(2)))  float  v2f;
typedef __attribute__((ext_vector_type(4)))  unsigned u32x4;
typedef __attribute__((ext_vector_type(8)))  int      i32x8;
typedef __attribute__((ext_vector_type(4)))  int      i32x4;

#if defined(__has_builtin)
#if __has_builtin(__builtin_amdgcn_wmma_f32_16x16x4_f32)
#define HAVE_WMMA_F32 1
#endif
#if __has_builtin(__builtin_amdgcn_tensor_load_to_lds) && \
    __has_builtin(__builtin_amdgcn_s_wait_tensorcnt)
#define HAVE_TDM 1
#endif
#endif

union BF16Frag {
    v16bf v;
    unsigned u[8];
    uint4 q[2];
};

// ---------------------------------------------------------------------------
// float -> bf16 (RNE) convert
// ---------------------------------------------------------------------------
__global__ void cvt_bf16_kernel(const float* __restrict__ src,
                                unsigned short* __restrict__ dst, int n) {
    int i = blockIdx.x * blockDim.x + threadIdx.x;
    if (i < n) {
        unsigned u = __float_as_uint(src[i]);
        unsigned r = (u + 0x7FFFu + ((u >> 16) & 1u)) >> 16;
        dst[i] = (unsigned short)r;
    }
}

// ---------------------------------------------------------------------------
// k = x @ map_w^T + map_b   (M=TB, N=RD, K=CIN)  -- f32 WMMA 16x16x4
// ---------------------------------------------------------------------------
__global__ void kproj_kernel(const float* __restrict__ x,
                             const float* __restrict__ mw,
                             const float* __restrict__ mb,
                             float* __restrict__ kout) {
    const int m0 = blockIdx.x * 16, n0 = blockIdx.y * 16;
    const int lane = threadIdx.x & 31;
    const int r = lane & 15, hi = lane >> 4;
#if HAVE_WMMA_F32
    v8f c = {};
    #pragma unroll 4
    for (int kk = 0; kk < CIN; kk += 4) {
        const int ks = kk + hi * 2;
        v2f a, b;
        a.x = x[(m0 + r) * CIN + ks];
        a.y = x[(m0 + r) * CIN + ks + 1];
        b.x = mw[(n0 + r) * CIN + ks];     // B[k][n] = map_w[n][k]
        b.y = mw[(n0 + r) * CIN + ks + 1];
        c = __builtin_amdgcn_wmma_f32_16x16x4_f32(false, a, false, b,
                                                  (short)0, c, false, false);
    }
    #pragma unroll
    for (int j = 0; j < 8; j++) {
        const int M = j + hi * 8, N = r;
        kout[(m0 + M) * RD + (n0 + N)] = c[j] + mb[n0 + N];
    }
#else
    for (int j = lane; j < 256; j += 32) {
        const int M = j >> 4, N = j & 15;
        float s = mb[n0 + N];
        for (int kk = 0; kk < CIN; kk++)
            s += x[(m0 + M) * CIN + kk] * mw[(n0 + N) * CIN + kk];
        kout[(m0 + M) * RD + (n0 + N)] = s;
    }
#endif
}

// ---------------------------------------------------------------------------
// dots = k @ centroids^T   (M=TB, N=NE, K=RD) -- f32 WMMA 16x16x4
// ---------------------------------------------------------------------------
__global__ void kdots_kernel(const float* __restrict__ k,
                             const float* __restrict__ cent,
                             float* __restrict__ dots) {
    const int m0 = blockIdx.x * 16, n0 = blockIdx.y * 16;
    const int lane = threadIdx.x & 31;
    const int r = lane & 15, hi = lane >> 4;
#if HAVE_WMMA_F32
    v8f c = {};
    #pragma unroll 4
    for (int kk = 0; kk < RD; kk += 4) {
        const int ks = kk + hi * 2;
        v2f a, b;
        a.x = k[(m0 + r) * RD + ks];
        a.y = k[(m0 + r) * RD + ks + 1];
        b.x = cent[(n0 + r) * RD + ks];
        b.y = cent[(n0 + r) * RD + ks + 1];
        c = __builtin_amdgcn_wmma_f32_16x16x4_f32(false, a, false, b,
                                                  (short)0, c, false, false);
    }
    #pragma unroll
    for (int j = 0; j < 8; j++) {
        const int M = j + hi * 8, N = r;
        dots[(m0 + M) * NE + (n0 + N)] = c[j];
    }
#else
    for (int j = lane; j < 256; j += 32) {
        const int M = j >> 4, N = j & 15;
        float s = 0.f;
        for (int kk = 0; kk < RD; kk++)
            s += k[(m0 + M) * RD + kk] * cent[(n0 + N) * RD + kk];
        dots[(m0 + M) * NE + (n0 + N)] = s;
    }
#endif
}

// ---------------------------------------------------------------------------
// csum[e] = sum(centroids[e]^2)
// ---------------------------------------------------------------------------
__global__ void csum_kernel(const float* __restrict__ cent,
                            float* __restrict__ csum) {
    int e = threadIdx.x;
    if (e < NE) {
        float s = 0.f;
        for (int i = 0; i < RD; i++) { float v = cent[e * RD + i]; s += v * v; }
        csum[e] = s;
    }
}

// ---------------------------------------------------------------------------
// threefry2x32 (JAX PRNG, key = PRNGKey(42) = (0,42)) -> gumbel noise
// ---------------------------------------------------------------------------
#define TF_ROUND(x0, x1, rr) { x0 += x1; x1 = (x1 << rr) | (x1 >> (32 - rr)); x1 ^= x0; }

__device__ __forceinline__ float gumbel_noise(unsigned w) {
    const unsigned NH = (TB * NE) / 2;   // 65536
    const unsigned p = (w < NH) ? w : (w - NH);
    const unsigned ks0 = 0u, ks1 = 42u;
    const unsigned ks2 = 0x1BD11BDAu ^ ks0 ^ ks1;
    unsigned x0 = p + ks0;
    unsigned x1 = (p + NH) + ks1;
    TF_ROUND(x0, x1, 13) TF_ROUND(x0, x1, 15) TF_ROUND(x0, x1, 26) TF_ROUND(x0, x1, 6)
    x0 += ks1; x1 += ks2 + 1u;
    TF_ROUND(x0, x1, 17) TF_ROUND(x0, x1, 29) TF_ROUND(x0, x1, 16) TF_ROUND(x0, x1, 24)
    x0 += ks2; x1 += ks0 + 2u;
    TF_ROUND(x0, x1, 13) TF_ROUND(x0, x1, 15) TF_ROUND(x0, x1, 26) TF_ROUND(x0, x1, 6)
    x0 += ks0; x1 += ks1 + 3u;
    TF_ROUND(x0, x1, 17) TF_ROUND(x0, x1, 29) TF_ROUND(x0, x1, 16) TF_ROUND(x0, x1, 24)
    x0 += ks1; x1 += ks2 + 4u;
    TF_ROUND(x0, x1, 13) TF_ROUND(x0, x1, 15) TF_ROUND(x0, x1, 26) TF_ROUND(x0, x1, 6)
    x0 += ks2; x1 += ks0 + 5u;
    const unsigned bits = (w < NH) ? x0 : x1;
    const float TINY = 1.1754943508222875e-38f;
    float f = __uint_as_float((bits >> 9) | 0x3F800000u) - 1.0f;  // [0,1)
    float u = f * (1.0f - TINY) + TINY;
    u = fmaxf(u, TINY);
    return -__logf(-__logf(u));
}

// ---------------------------------------------------------------------------
// per-token routing: distances -> gumbel softmax -> resp; quantize; losses
// ---------------------------------------------------------------------------
__global__ void route_kernel(const float* __restrict__ k,
                             const float* __restrict__ dots,
                             const float* __restrict__ csum,
                             const float* __restrict__ cent,
                             float* __restrict__ resp,
                             float* __restrict__ importance,
                             float* __restrict__ mse_acc) {
    const int t = blockIdx.x * blockDim.x + threadIdx.x;
    if (t >= TB) return;
    float kr[RD];
    float ksum = 0.f;
    #pragma unroll 8
    for (int i = 0; i < RD; i++) { kr[i] = k[t * RD + i]; ksum += kr[i] * kr[i]; }

    float lg[NE];
    float mx = -3.4e38f;
    for (int e = 0; e < NE; e++) {
        const float d = ksum + csum[e] - 2.f * dots[t * NE + e];
        const float g = gumbel_noise((unsigned)(t * NE + e));
        const float l = (-d + g) / TAU;
        lg[e] = l;
        mx = fmaxf(mx, l);
    }
    float s = 0.f;
    for (int e = 0; e < NE; e++) { float w = __expf(lg[e] - mx); lg[e] = w; s += w; }
    const float inv = 1.f / s;
    for (int e = 0; e < NE; e++) {
        const float r = lg[e] * inv;
        lg[e] = r;
        resp[t * NE + e] = r;
        atomicAdd(&importance[e], r);
    }
    float mse = 0.f;
    for (int rd = 0; rd < RD; rd++) {
        float q = 0.f;
        #pragma unroll 8
        for (int e = 0; e < NE; e++) q += lg[e] * cent[e * RD + rd];
        const float diff = kr[rd] - q;
        mse += diff * diff;
    }
    atomicAdd(mse_acc, mse);
}

// ---------------------------------------------------------------------------
// out[t,o] = sum_e resp[t,e] * (x[t,:] . pw_w[e,o,:]) + pw_B[o]
// bf16 WMMA 16x16x32, f32 accumulate; B staged into LDS by the Tensor Data
// Mover (double-buffered on TENSORcnt), with D# row padding for bank skew.
// block = 128 threads (4 waves): 16-token tile x 4 N-tiles (64 outputs)
// grid (TB/16, COUT/64)
// ---------------------------------------------------------------------------
#define XS_STRIDE 272          // padded x row stride (halfs)
#define ROWB 544               // padded B row stride in LDS (512 data + 32 pad)
#define BBUF_BYTES (64 * ROWB) // 34816 per buffer

struct CombineSmem {
    char bbuf[2][BBUF_BYTES];          // offset 0 / 34816 from LDS base
    unsigned short xs[16 * XS_STRIDE]; // x tile
    float rs[16 * NE];                 // resp tile
};

#if HAVE_TDM
// DMA one expert tile pw_w[e][o0..o0+64)[0..256) (bf16) into LDS at lds_off.
// 2D tile: dim0=256 elem rows, 64 rows, 2B elements; pad 32B per 512B row.
__device__ __forceinline__ void tdm_load_expert(const unsigned short* gsrc,
                                                unsigned lds_off) {
    const unsigned long long ga = (unsigned long long)(uintptr_t)gsrc;
    u32x4 g0;
    g0[0] = 1u;                                   // count=1, user descriptor
    g0[1] = lds_off;                              // LDS byte address
    g0[2] = (unsigned)(ga & 0xFFFFFFFFu);         // global addr [31:0]
    g0[3] = (unsigned)((ga >> 32) & 0x1FFFFFFu)   // global addr [56:32]
            | (2u << 30);                         // type = 2 (image)
    i32x8 g1;
    g1[0] = (int)((1u << 16)                      // data_size = 2 bytes
                  | (1u << 20)                    // pad_enable
                  | (6u << 22)                    // pad_interval: 512B
                  | (7u << 25));                  // pad_amount: 32B
    g1[1] = (int)(256u << 16);                    // tensor_dim0 = 256
    g1[2] = (int)(4096u << 16);                   // tensor_dim1 = 4096
    g1[3] = (int)(256u << 16);                    // tile_dim0 = 256
    g1[4] = 64;                                   // tile_dim1 = 64
    g1[5] = 256;                                  // tensor_dim0_stride = 256
    g1[6] = 0;
    g1[7] = 0;
    i32x4 z4; z4[0] = 0; z4[1] = 0; z4[2] = 0; z4[3] = 0;
    i32x8 z8; z8[0] = 0; z8[1] = 0; z8[2] = 0; z8[3] = 0;
    z8[4] = 0; z8[5] = 0; z8[6] = 0; z8[7] = 0;
    // 6-arg form (clang-23 / therock headers): groups 0..3, spare, cpol
    __builtin_amdgcn_tensor_load_to_lds(g0, g1, z4, z4, z8, 0);
}
#endif

__global__ void __launch_bounds__(128) combine_kernel(
        const unsigned short* __restrict__ xbf,
        const unsigned short* __restrict__ pwbf,
        const float* __restrict__ resp,
        const float* __restrict__ pwB,
        float* __restrict__ out) {
    __shared__ CombineSmem sm;

    const int m0 = blockIdx.x * 16;
    const int wave = threadIdx.x >> 5;
    const int lane = threadIdx.x & 31;
    const int oblk = blockIdx.y * 64;          // first output of this block
    const int n0 = oblk + wave * 16;           // this wave's output tile

    // stage x tile (16 x 256 bf16) and resp tile (16 x 64 f32) into LDS
    for (int i = threadIdx.x; i < 16 * CIN; i += 128) {
        const int rr = i >> 8, cc = i & 255;
        sm.xs[rr * XS_STRIDE + cc] = xbf[(m0 + rr) * CIN + cc];
    }
    for (int i = threadIdx.x; i < 16 * NE; i += 128)
        sm.rs[i] = resp[m0 * NE + i];

#if HAVE_TDM
    // prologue: wave 0 DMAs expert 0's B tile into buffer 0
    if (wave == 0) {
        tdm_load_expert(pwbf + (size_t)oblk * CIN, 0u);
        __builtin_amdgcn_s_wait_tensorcnt((short)0);
    }
#endif
    __syncthreads();

    const int r = lane & 15, hi = lane >> 4;

    // A fragments for all 8 K-chunks (expert-invariant -> hoisted)
    BF16Frag afrag[8];
    #pragma unroll
    for (int c2 = 0; c2 < 8; c2++) {
        const int kbase = c2 * 32;
        #pragma unroll
        for (int w = 0; w < 8; w++) {
            const int bk = ((w < 4) ? 0 : 16) + hi * 8 + (w & 3) * 2;
            afrag[c2].u[w] =
                *(const unsigned*)&sm.xs[r * XS_STRIDE + kbase + bk];
        }
    }

    const int brow = wave * 16 + r;            // row in staged B buffer
    v8f acc = {};
    int cur = 0;
    for (int e = 0; e < NE; e++) {
#if HAVE_TDM
        if (wave == 0 && e + 1 < NE)
            tdm_load_expert(pwbf + (size_t)(e + 1) * (COUT * CIN)
                                 + (size_t)oblk * CIN,
                            (unsigned)((cur ^ 1) * BBUF_BYTES));
#else
        {   // synchronous cooperative copy of expert e's tile into buffer 0
            const char* src = (const char*)(pwbf + (size_t)e * (COUT * CIN)
                                                 + (size_t)oblk * CIN);
            for (int i = threadIdx.x; i < 64 * 32; i += 128) {
                const int row = i >> 5, seg = i & 31;     // 32 x 16B per row
                *(uint4*)(sm.bbuf[0] + row * ROWB + seg * 16) =
                    *(const uint4*)(src + row * 512 + seg * 16);
            }
            __syncthreads();
        }
#endif
        const unsigned short* bl =
            (const unsigned short*)(sm.bbuf[cur] + brow * ROWB);
        v8f ce = {};
        #pragma unroll
        for (int c2 = 0; c2 < 8; c2++) {
            BF16Frag bfrag;
            const unsigned short* q = bl + c2 * 32 + hi * 8;
            bfrag.q[0] = *(const uint4*)q;
            bfrag.q[1] = *(const uint4*)(q + 16);
            ce = __builtin_amdgcn_wmma_f32_16x16x32_bf16(
                false, afrag[c2].v, false, bfrag.v, (short)0, ce, false, false);
        }
        #pragma unroll
        for (int j = 0; j < 8; j++) {
            const int M = j + hi * 8;
            acc[j] += sm.rs[M * NE + e] * ce[j];
        }
#if HAVE_TDM
        if (wave == 0)
            __builtin_amdgcn_s_wait_tensorcnt((short)0);
        __syncthreads();
        cur ^= 1;
#else
        __syncthreads();
#endif
    }

    const float bias = pwB[n0 + r];
    #pragma unroll
    for (int j = 0; j < 8; j++) {
        const int M = j + hi * 8;
        out[(size_t)(m0 + M) * COUT + n0 + r] = acc[j] + bias;
    }
}

// ---------------------------------------------------------------------------
// finalize scalar loss (commitment == kmeans in forward; both = mse mean)
// ---------------------------------------------------------------------------
__global__ void loss_kernel(const float* __restrict__ importance,
                            const float* __restrict__ mse_acc,
                            float* __restrict__ loss_out) {
    float sum = 0.f, sumsq = 0.f;
    for (int e = 0; e < NE; e++) {
        const float v = importance[e];
        sum += v; sumsq += v * v;
    }
    const float mean = sum / (float)NE;
    const float var = (sumsq - (float)NE * mean * mean) / (float)(NE - 1);
    const float imp_loss = sqrtf(fmaxf(var, 0.f)) / mean;
    const float mse = mse_acc[0] / (float)(TB * RD);
    loss_out[0] = (COMMIT + KMEANS) * mse + IMP * imp_loss;
}

// ---------------------------------------------------------------------------
// launch
// ---------------------------------------------------------------------------
extern "C" void kernel_launch(void* const* d_in, const int* in_sizes, int n_in,
                              void* d_out, int out_size, void* d_ws, size_t ws_size,
                              hipStream_t stream) {
    const float* x    = (const float*)d_in[0];   // (TB,CIN)
    const float* mw   = (const float*)d_in[1];   // (RD,CIN)
    const float* mb   = (const float*)d_in[2];   // (RD,)
    const float* cent = (const float*)d_in[3];   // (NE,RD)
    const float* pww  = (const float*)d_in[4];   // (NE, COUT*CIN)
    const float* pwB  = (const float*)d_in[5];   // (1,1,COUT)
    float* out  = (float*)d_out;                 // TB*COUT floats
    float* loss = out + (size_t)TB * COUT;       // +1 float

    char* w = (char*)d_ws;
    float* kbuf = (float*)(w);                               // 512 KB
    float* dots = (float*)(w + (512u << 10));                // 512 KB
    float* resp = (float*)(w + (1024u << 10));               // 512 KB
    float* csum = (float*)(w + (1536u << 10));               // 256 B
    float* impo = (float*)(w + (1536u << 10) + 256);         // 256 B
    float* msea = (float*)(w + (1536u << 10) + 512);         // 256 B
    unsigned short* xbf  = (unsigned short*)(w + (1536u << 10) + 1024);              // 1 MB
    unsigned short* pwbf = (unsigned short*)(w + (1536u << 10) + 1024 + (1u << 20)); // 8.4 MB

    // zero the atomics (importance + mse) each call (graph-capture safe)
    (void)hipMemsetAsync(impo, 0, 512, stream);

    // bf16 conversions
    {
        const int n1 = TB * CIN;
        cvt_bf16_kernel<<<(n1 + 255) / 256, 256, 0, stream>>>(x, xbf, n1);
        const int n2 = NE * COUT * CIN;
        cvt_bf16_kernel<<<(n2 + 255) / 256, 256, 0, stream>>>(pww, pwbf, n2);
    }
    // router: k projection + dots + csum
    kproj_kernel<<<dim3(TB / 16, RD / 16), 32, 0, stream>>>(x, mw, mb, kbuf);
    kdots_kernel<<<dim3(TB / 16, NE / 16), 32, 0, stream>>>(kbuf, cent, dots);
    csum_kernel<<<1, 64, 0, stream>>>(cent, csum);
    // gumbel softmax + quantization losses
    route_kernel<<<TB / 256, 256, 0, stream>>>(kbuf, dots, csum, cent,
                                               resp, impo, msea);
    // heavy expert combine (TDM double-buffered B stage)
    combine_kernel<<<dim3(TB / 16, COUT / 64), 128, 0, stream>>>(
        xbf, pwbf, resp, pwB, out);
    // scalar loss
    loss_kernel<<<1, 1, 0, stream>>>(impo, msea, loss);
}
